// GQA_SA_83880711291011
// MI455X (gfx1250) — matrix operand order repaired and checked by reading, not compiled, not run on hardware
//
#include <hip/hip_runtime.h>
#include <hip/hip_bf16.h>

// ---------------------------------------------------------------------------
// GQA self-attention for MI455X (gfx1250), all GEMM stages on
// v_wmma_f32_16x16x32_f16 (f16 inputs, f32 accumulation), async LDS staging.
// ---------------------------------------------------------------------------

typedef _Float16 h16;
typedef __attribute__((ext_vector_type(16))) _Float16 v16h;
typedef __attribute__((ext_vector_type(8)))  _Float16 v8h;
typedef __attribute__((ext_vector_type(8)))  float    v8f;

constexpr int B_C   = 2;
constexpr int S_C   = 2048;
constexpr int D_C   = 2048;
constexpr int HQ_C  = 16;
constexpr int HKV_C = 4;
constexpr int DH_C  = 128;
constexpr float SCALE_C = 0.08838834764831845f;   // 128^-0.5
constexpr float MASK_C  = -1.0e9f;

#if defined(__gfx1250__) && __has_builtin(__builtin_amdgcn_global_load_async_to_lds_b128)
#define HAVE_ASYNC_LDS 1
// Builtin expects pointer-to-<4 x i32> in global (AS1) / LDS (AS3) address spaces
// (diagnosed signature: "__attribute__((vector_size(4*sizeof(int)))) int __device__ *").
typedef int v4i_t __attribute__((vector_size(4 * sizeof(int))));
typedef __attribute__((address_space(1))) v4i_t* gv4_p;
typedef __attribute__((address_space(3))) v4i_t* lv4_p;
#else
#define HAVE_ASYNC_LDS 0
#endif

static __device__ __forceinline__ void wait_async_zero() {
#if HAVE_ASYNC_LDS
#if __has_builtin(__builtin_amdgcn_s_wait_asynccnt)
  __builtin_amdgcn_s_wait_asynccnt(0);
#else
  asm volatile("s_wait_asynccnt 0" ::: "memory");
#endif
#endif
}

static __device__ __forceinline__ v8f wmma_f16(v16h a, v16h b, v8f c) {
  return __builtin_amdgcn_wmma_f32_16x16x32_f16(
      /*neg_a=*/false, a, /*neg_b=*/false, b,
      /*c_mod=*/(short)0, c, /*reuse_a=*/false, /*reuse_b=*/false);
}

// A-matrix fragment (16x32 f16, MxK), per ISA layout:
//   lanes 0-15 : m=lane,   halves 0-7 -> k=0..7,  halves 8-15 -> k=16..23
//   lanes 16-31: m=lane-16,halves 0-7 -> k=8..15, halves 8-15 -> k=24..31
static __device__ __forceinline__ v16h frag_a(const h16* base, int ld, int k0, int lane) {
  const int row = lane & 15, hl = lane >> 4;
  const h16* p = base + row * ld + k0 + 8 * hl;
  v8h lo = *(const v8h*)p;
  v8h hi = *(const v8h*)(p + 16);
  return __builtin_shufflevector(lo, hi, 0,1,2,3,4,5,6,7,8,9,10,11,12,13,14,15);
}

// B-matrix fragment (32x16 f16, KxN): lane L holds n=L%16, halves 0-7 -> k=8*(L/16)+h, halves 8-15 -> k=16+8*(L/16)+(h-8).
// base is row-major [n][k] with leading dim ld.
static __device__ __forceinline__ v16h frag_b(const h16* base, int ld, int k0, int lane) {
  const int n = lane & 15, hl = lane >> 4;
  const h16* p = base + n * ld + k0 + 8 * hl;
  v8h lo = *(const v8h*)p;
  v8h hi = *(const v8h*)(p + 16);
  return __builtin_shufflevector(lo, hi, 0,1,2,3,4,5,6,7,8,9,10,11,12,13,14,15);
}

// ---------------------------------------------------------------------------
// GEMM: C[M,N] = A[M,K] * W[N,K]^T + bias[N]
// Block tile 128x128, K staged 64 at a time (16 WMMAs/wave between barriers),
// 256 threads = 8 waves (4x2), each wave computes 32x64 = 2x4 fragments.
// ---------------------------------------------------------------------------
template <bool AF16>
__global__ __launch_bounds__(256)
void gqa_gemm_wmma(const void* __restrict__ Ap, const float* __restrict__ W,
                   const float* __restrict__ bias, float* __restrict__ C,
                   int M, int N, int K) {
  __shared__ h16 As[128][72];   // 64 k-cols used; 144B row stride (16B aligned)
  __shared__ h16 Bs[128][72];

  const int tid  = threadIdx.x;
  const int lane = tid & 31;
  const int wave = tid >> 5;
  const int wm = wave >> 1;          // 0..3
  const int wn = wave & 1;           // 0..1
  const int rowBase = blockIdx.y * 128;
  const int colBase = blockIdx.x * 128;

  v8f acc[2][4];
#pragma unroll
  for (int fm = 0; fm < 2; ++fm)
#pragma unroll
    for (int fn = 0; fn < 4; ++fn) acc[fm][fn] = v8f{};

  const int ldRow = tid >> 1;          // 0..127
  const int ldSeg = (tid & 1) * 32;    // 0 or 32

  for (int kt = 0; kt < K; kt += 64) {
    // ---- stage A tile into LDS (convert fp32->f16 if needed) ----
    if (AF16) {
      const h16* src = (const h16*)Ap + (size_t)(rowBase + ldRow) * K + kt + ldSeg;
      const int4* s4 = (const int4*)src;
      int4* d4 = (int4*)&As[ldRow][ldSeg];
      d4[0] = s4[0]; d4[1] = s4[1]; d4[2] = s4[2]; d4[3] = s4[3];
    } else {
      const float* src = (const float*)Ap + (size_t)(rowBase + ldRow) * K + kt + ldSeg;
#pragma unroll
      for (int j = 0; j < 32; j += 4) {
        float4 f = *(const float4*)(src + j);
        As[ldRow][ldSeg + j + 0] = (h16)f.x;
        As[ldRow][ldSeg + j + 1] = (h16)f.y;
        As[ldRow][ldSeg + j + 2] = (h16)f.z;
        As[ldRow][ldSeg + j + 3] = (h16)f.w;
      }
    }
    // ---- stage B tile (weight rows, fp32->f16) ----
    const float* wsrc = W + (size_t)(colBase + ldRow) * K + kt + ldSeg;
#pragma unroll
    for (int j = 0; j < 32; j += 4) {
      float4 f = *(const float4*)(wsrc + j);
      Bs[ldRow][ldSeg + j + 0] = (h16)f.x;
      Bs[ldRow][ldSeg + j + 1] = (h16)f.y;
      Bs[ldRow][ldSeg + j + 2] = (h16)f.z;
      Bs[ldRow][ldSeg + j + 3] = (h16)f.w;
    }
    if (kt + 64 < K) {
      __builtin_prefetch(wsrc + 64, 0, 3);          // global_prefetch_b8
      if (!AF16)
        __builtin_prefetch((const float*)Ap + (size_t)(rowBase + ldRow) * K + kt + 64 + ldSeg, 0, 3);
    }
    __syncthreads();

#pragma unroll
    for (int kc = 0; kc < 64; kc += 32) {
      v16h a[2], b[4];
#pragma unroll
      for (int fm = 0; fm < 2; ++fm)
        a[fm] = frag_a(&As[32 * wm + 16 * fm][0], 72, kc, lane);
#pragma unroll
      for (int fn = 0; fn < 4; ++fn)
        b[fn] = frag_b(&Bs[64 * wn + 16 * fn][0], 72, kc, lane);
#pragma unroll
      for (int fm = 0; fm < 2; ++fm)
#pragma unroll
        for (int fn = 0; fn < 4; ++fn)
          acc[fm][fn] = wmma_f16(a[fm], b[fn], acc[fm][fn]);
    }
    __syncthreads();
  }

  // ---- epilogue: bias add, fp32 store ----
  const int hl = lane >> 4, nl = lane & 15;
#pragma unroll
  for (int fm = 0; fm < 2; ++fm) {
#pragma unroll
    for (int fn = 0; fn < 4; ++fn) {
      const int col = colBase + 64 * wn + 16 * fn + nl;
      const float bv = bias[col];
#pragma unroll
      for (int i = 0; i < 8; ++i) {
        const int row = rowBase + 32 * wm + 16 * fm + i + 8 * hl;
        C[(size_t)row * N + col] = acc[fm][fn][i] + bv;
      }
    }
  }
}

// ---------------------------------------------------------------------------
// RoPE + pack fp32 projections into head-major f16 tensors.
// ---------------------------------------------------------------------------
__global__ void gqa_rope_pack(const float* __restrict__ qws, const float* __restrict__ kws,
                              const float* __restrict__ vws, const float* __restrict__ rope_freq,
                              h16* __restrict__ Qh, h16* __restrict__ Kh, h16* __restrict__ Vh) {
  const int gid = blockIdx.x * blockDim.x + threadIdx.x;
  const int TOT = B_C * S_C * (HQ_C + 2 * HKV_C) * (DH_C / 2);
  if (gid >= TOT) return;
  const int pair = gid & 63;
  int t = gid >> 6;
  const int head = t % 24;
  t /= 24;
  const int s = t % S_C;
  const int b = t / S_C;

  if (head < HQ_C) {
    const float* src = qws + ((size_t)(b * S_C + s)) * D_C + head * DH_C + 2 * pair;
    const float fr = rope_freq[s * 64 + pair];
    const float c = __cosf(fr), sn = __sinf(fr);
    const float xr = src[0], xi = src[1];
    h16* dst = Qh + (((size_t)(b * HQ_C + head)) * S_C + s) * DH_C + 2 * pair;
    dst[0] = (h16)(xr * c - xi * sn);
    dst[1] = (h16)(xr * sn + xi * c);
  } else if (head < HQ_C + HKV_C) {
    const int kh = head - HQ_C;
    const float* src = kws + ((size_t)(b * S_C + s)) * (HKV_C * DH_C) + kh * DH_C + 2 * pair;
    const float fr = rope_freq[s * 64 + pair];
    const float c = __cosf(fr), sn = __sinf(fr);
    const float xr = src[0], xi = src[1];
    h16* dst = Kh + (((size_t)(b * HKV_C + kh)) * S_C + s) * DH_C + 2 * pair;
    dst[0] = (h16)(xr * c - xi * sn);
    dst[1] = (h16)(xr * sn + xi * c);
  } else {
    const int vh = head - HQ_C - HKV_C;
    const float* src = vws + ((size_t)(b * S_C + s)) * (HKV_C * DH_C) + vh * DH_C + 2 * pair;
    h16* dst = Vh + (((size_t)(b * HKV_C + vh)) * S_C + s) * DH_C + 2 * pair;
    dst[0] = (h16)src[0];
    dst[1] = (h16)src[1];
  }
}

// ---------------------------------------------------------------------------
// Flash attention (causal, online softmax). Grid: (S/64, HQ, B); 128 threads.
// Wave w owns query rows q0+16w..+15. KV staged in 64-key LDS tiles (K via
// async-to-LDS when available, V transposed). All tiles up to the block's
// diagonal intersect every wave, so there is no divergent skip around WMMA.
// ---------------------------------------------------------------------------
__global__ __launch_bounds__(128)
void gqa_flash_attn(const h16* __restrict__ Qh, const h16* __restrict__ Kh,
                    const h16* __restrict__ Vh, h16* __restrict__ attn_h) {
  __shared__ h16 Kt[64][136];       // [key][d], 272B rows (16B aligned)
  __shared__ h16 VtT[128][72];      // [d][key], 144B rows
  __shared__ h16 Pst[4][16][72];    // per-wave P staging (C-layout -> A-layout)

  const int tid  = threadIdx.x;
  const int lane = tid & 31;
  const int w    = tid >> 5;
  const int hl   = lane >> 4, nl = lane & 15;
  const int q0 = blockIdx.x * 64;
  const int h  = blockIdx.y;
  const int b  = blockIdx.z;
  const int hkv = h >> 2;                     // GROUPS = 4
  const int qrw = q0 + 16 * w;                // first query row of this wave

  const h16* Qbase = Qh + ((size_t)(b * HQ_C + h) * S_C + qrw) * DH_C;
  v16h qa[4];
#pragma unroll
  for (int c = 0; c < 4; ++c) qa[c] = frag_a(Qbase, DH_C, 32 * c, lane);

  v8f o[8];
#pragma unroll
  for (int nt = 0; nt < 8; ++nt) o[nt] = v8f{};
  float rm[8], rl[8];
#pragma unroll
  for (int i = 0; i < 8; ++i) { rm[i] = -__builtin_inff(); rl[i] = 0.0f; }

  const h16* Kbase = Kh + (size_t)(b * HKV_C + hkv) * S_C * DH_C;
  const h16* Vbase = Vh + (size_t)(b * HKV_C + hkv) * S_C * DH_C;

  const int ldRow = tid >> 1;          // 0..63: key row
  const int ldSeg = (tid & 1) * 64;    // 0 or 64: d segment

  const int nTiles = blockIdx.x + 1;   // 64-key tiles covering keys [0, q0+64)
  for (int t = 0; t < nTiles; ++t) {
    const int kb = t * 64;
    __syncthreads();   // previous tile fully consumed before overwrite
    // ---- K tile: 64 keys x 128 dims ----
    const h16* ksrc = Kbase + (size_t)(kb + ldRow) * DH_C + ldSeg;
    h16* kdst = &Kt[ldRow][ldSeg];
#if HAVE_ASYNC_LDS
#pragma unroll
    for (int j = 0; j < 8; ++j)
      __builtin_amdgcn_global_load_async_to_lds_b128(
          (gv4_p)(uintptr_t)(ksrc + 8 * j), (lv4_p)(kdst + 8 * j), 0, 0);
#else
    {
      const int4* s4 = (const int4*)ksrc;
      int4* d4 = (int4*)kdst;
#pragma unroll
      for (int j = 0; j < 8; ++j) d4[j] = s4[j];
    }
#endif
    // ---- V tile transposed into [d][key] ----
    {
      const h16* vsrc = Vbase + (size_t)(kb + ldRow) * DH_C + ldSeg;
#pragma unroll
      for (int j = 0; j < 64; ++j) VtT[ldSeg + j][ldRow] = vsrc[j];
    }
    wait_async_zero();
    __syncthreads();

    // ---- scores: four 16-key sub-tiles, 4 WMMAs each over DH=128 ----
    v8f sc[4];
#pragma unroll
    for (int n0 = 0; n0 < 4; ++n0) {
      v8f a = v8f{};
#pragma unroll
      for (int c = 0; c < 4; ++c) {
        v16h bk = frag_b(&Kt[16 * n0][0], 136, 32 * c, lane);
        a = wmma_f16(qa[c], bk, a);
      }
      const int key = kb + 16 * n0 + nl;
#pragma unroll
      for (int i = 0; i < 8; ++i) {
        const int qr = qrw + i + 8 * hl;
        a[i] = (a[i] + (key > qr ? MASK_C : 0.0f)) * SCALE_C;
      }
      sc[n0] = a;
    }

    // ---- online softmax (row stats replicated across each 16-lane half) ----
    float corr[8];
#pragma unroll
    for (int i = 0; i < 8; ++i) {
      float mx = fmaxf(fmaxf(sc[0][i], sc[1][i]), fmaxf(sc[2][i], sc[3][i]));
      mx = fmaxf(mx, __shfl_xor(mx, 1, 32));
      mx = fmaxf(mx, __shfl_xor(mx, 2, 32));
      mx = fmaxf(mx, __shfl_xor(mx, 4, 32));
      mx = fmaxf(mx, __shfl_xor(mx, 8, 32));
      const float mnew = fmaxf(rm[i], mx);
      corr[i] = __expf(rm[i] - mnew);
      float rs = 0.0f;
#pragma unroll
      for (int n0 = 0; n0 < 4; ++n0) {
        const float p = __expf(sc[n0][i] - mnew);
        sc[n0][i] = p;
        rs += p;
      }
      rs += __shfl_xor(rs, 1, 32);
      rs += __shfl_xor(rs, 2, 32);
      rs += __shfl_xor(rs, 4, 32);
      rs += __shfl_xor(rs, 8, 32);
      rl[i] = rl[i] * corr[i] + rs;
      rm[i] = mnew;
    }
#pragma unroll
    for (int nt = 0; nt < 8; ++nt)
#pragma unroll
      for (int i = 0; i < 8; ++i) o[nt][i] *= corr[i];

    // ---- P: C-layout -> A-layout via per-wave LDS staging ----
#pragma unroll
    for (int n0 = 0; n0 < 4; ++n0)
#pragma unroll
      for (int i = 0; i < 8; ++i)
        Pst[w][i + 8 * hl][16 * n0 + nl] = (h16)sc[n0][i];
    asm volatile("s_wait_dscnt 0" ::: "memory");
    v16h pa[2];
#pragma unroll
    for (int c = 0; c < 2; ++c) pa[c] = frag_a(&Pst[w][0][0], 72, 32 * c, lane);

    // ---- O += P @ V : 8 d-tiles x 2 key-chunks ----
#pragma unroll
    for (int nt = 0; nt < 8; ++nt) {
#pragma unroll
      for (int c = 0; c < 2; ++c) {
        v16h bv = frag_b(&VtT[16 * nt][0], 72, 32 * c, lane);
        o[nt] = wmma_f16(pa[c], bv, o[nt]);
      }
    }
  }

  // ---- normalize and store f16 in [b, s, h*DH + d] layout ----
#pragma unroll
  for (int i = 0; i < 8; ++i) {
    const float inv = 1.0f / rl[i];
    const size_t row = (size_t)(b * S_C + qrw + i + 8 * hl);
#pragma unroll
    for (int nt = 0; nt < 8; ++nt) {
      attn_h[row * D_C + h * DH_C + 16 * nt + nl] = (h16)(o[nt][i] * inv);
    }
  }
}

// ---------------------------------------------------------------------------
extern "C" void kernel_launch(void* const* d_in, const int* in_sizes, int n_in,
                              void* d_out, int out_size, void* d_ws, size_t ws_size,
                              hipStream_t stream) {
  const float* x   = (const float*)d_in[0];
  const float* rf  = (const float*)d_in[1];
  // d_in[2] = mask (causal mask recomputed in-kernel)
  const float* wq  = (const float*)d_in[3];
  const float* wqb = (const float*)d_in[4];
  const float* wk  = (const float*)d_in[5];
  const float* wkb = (const float*)d_in[6];
  const float* wv  = (const float*)d_in[7];
  const float* wvb = (const float*)d_in[8];
  const float* wo  = (const float*)d_in[9];
  const float* wob = (const float*)d_in[10];
  float* out = (float*)d_out;

  const int M  = B_C * S_C;          // 4096
  const int NK = HKV_C * DH_C;       // 512

  char* ws = (char*)d_ws;
  size_t off = 0;
  float* qws = (float*)(ws + off); off += (size_t)M * D_C * 4;     // 33.5 MB
  float* kws = (float*)(ws + off); off += (size_t)M * NK * 4;      //  8.4 MB
  float* vws = (float*)(ws + off); off += (size_t)M * NK * 4;      //  8.4 MB
  h16* Qh = (h16*)(ws + off); off += (size_t)M * D_C * 2;          // 16.8 MB
  h16* Kh = (h16*)(ws + off); off += (size_t)M * NK * 2;           //  4.2 MB
  h16* Vh = (h16*)(ws + off); off += (size_t)M * NK * 2;           //  4.2 MB
  h16* attn_h = (h16*)qws;   // alias: qws fully consumed before attention writes

  dim3 blk(256);
  gqa_gemm_wmma<false><<<dim3(D_C / 128, M / 128), blk, 0, stream>>>(x, wq, wqb, qws, M, D_C, D_C);
  gqa_gemm_wmma<false><<<dim3(NK / 128, M / 128), blk, 0, stream>>>(x, wk, wkb, kws, M, NK, D_C);
  gqa_gemm_wmma<false><<<dim3(NK / 128, M / 128), blk, 0, stream>>>(x, wv, wvb, vws, M, NK, D_C);

  const int ropeTot = B_C * S_C * (HQ_C + 2 * HKV_C) * (DH_C / 2);
  gqa_rope_pack<<<(ropeTot + 255) / 256, 256, 0, stream>>>(qws, kws, vws, rf, Qh, Kh, Vh);

  gqa_flash_attn<<<dim3(S_C / 64, HQ_C, B_C), dim3(128), 0, stream>>>(Qh, Kh, Vh, attn_h);

  gqa_gemm_wmma<true><<<dim3(D_C / 128, M / 128), blk, 0, stream>>>(attn_h, wo, wob, out, M, D_C, D_C);
}
